// BaseModel_29557964931251
// MI455X (gfx1250) — matrix operand order repaired and checked by reading, not compiled, not run on hardware
//
#include <hip/hip_runtime.h>
#include <hip/hip_bf16.h>

// ---------------------------------------------------------------------------
// SPINN stack-LSTM on MI455X (gfx1250, wave32, WMMA f32_16x16x32_f16).
//
// Per step (B=128, H=300, TRK=64):
//   KA  lstm_in = [buf_h|s1_h|s2_h](128x960,f16) @ Wb(960x256)      -> f16
//   KB  gates   = [lstm_in|th_prev](128x320)     @ Wg(320x256), LSTM cell,
//                 logits = th_new @ W_trans + b  -> d_out
//   KC  a       = [s2_h|s1_h|th_new](128x704)    @ Wr(704x1536pad) + b_reduce
//   KD  TreeLSTM reduce elementwise + stack shift/reduce update.
//
// GEMM inner loops are register-blocked: per kt, one A fragment from LDS is
// reused across 4 accumulators (4 N-tiles), with B fragments streamed from
// pre-swizzled L2-resident f16 weight images (two b128 loads per fragment).
// This keeps the per-wave working set ~50 VGPRs and avoids scratch spills.
// ---------------------------------------------------------------------------

typedef __attribute__((ext_vector_type(16))) _Float16 v16h;
typedef __attribute__((ext_vector_type(8)))  _Float16 v8h;
typedef __attribute__((ext_vector_type(8)))  float    v8f;

#define Bsz     128
#define Sseq    256
#define Hdim    300
#define D2      600      // 2H
#define TRK     64
#define NSTEPS  511
#define SLOTS   257      // S+1 stack slots
#define OUTCOLS 1322     // 300 + 2*511

#define KB_  960   // bundle K  (3 x 320-padded H)
#define KG_  320   // gates  K  (256 + 64)
#define KR_  704   // reduce K  (320 + 320 + 64)
#define NB_  256   // bundle/gates N
#define NR_  1536  // reduce N padded to 96 tiles (guard-free B loads)
#define NRr  1500  // real reduce N (5*H)

static __device__ inline float sigm(float x) { return 1.0f / (1.0f + __expf(-x)); }

// A fragment (16x32 f16) from LDS row-major tile [16][Kpad].
// M = lane%16 ; K = (lane>=16 ? 16:0) + j (contiguous, matches swizzled B).
static __device__ inline v16h load_a_lds(const _Float16* Xa, int Kpad, int kt, int lane) {
    int m    = lane & 15;
    int base = (lane & 16) ? 16 : 0;
    const _Float16* p = Xa + m * Kpad + kt * 32 + base;
    v8h lo = *(const v8h*)p;          // K = kt*32+base   .. +7
    v8h hi = *(const v8h*)(p + 8);    // K = kt*32+base+8 .. +15
    v16h a;
#pragma unroll
    for (int j = 0; j < 8; ++j) { a[j] = lo[j]; a[j + 8] = hi[j]; }
    return a;
}

// B fragment (32x16 f16) from a weight image pre-swizzled so each lane's 16
// halves are contiguous: element (k = kt*32 + (lane>=16?16:0) + j, n = nt*16 + lane%16).
static __device__ inline v16h load_b_sw(const _Float16* W, int ntiles, int kt, int nt, int lane) {
    const _Float16* p = W + ((((size_t)kt * ntiles + nt) * 32) + lane) * 16;
    v8h lo = *(const v8h*)p;
    v8h hi = *(const v8h*)(p + 8);
    v16h b;
#pragma unroll
    for (int j = 0; j < 8; ++j) { b[j] = lo[j]; b[j + 8] = hi[j]; }
    return b;
}

// ------------------------------ init kernels -------------------------------

__global__ void k_init_state(int* sptr, int* bptr, float* th, float* tc) {
    int i = blockIdx.x * 256 + threadIdx.x;
    if (i < Bsz) { sptr[i] = 0; bptr[i] = 0; }
    if (i < Bsz * TRK) { th[i] = 0.0f; tc[i] = 0.0f; }
}

// Build swizzled f16 weight images for bundle (Wb) and gates (Wg).
__global__ void k_build_wbg(const float* W_buf, const float* W_s1, const float* W_s2,
                            const float* W_ih, const float* W_hh,
                            _Float16* WbSW, _Float16* WgSW) {
    const int total_b = (KB_ / 32) * (NB_ / 16) * 512;   // 245760
    const int total_g = (KG_ / 32) * (NB_ / 16) * 512;   //  81920
    for (int idx = blockIdx.x * 256 + threadIdx.x; idx < total_b + total_g;
         idx += gridDim.x * 256) {
        bool isB = idx < total_b;
        int id   = isB ? idx : idx - total_b;
        int j    = id & 15;
        int lane = (id >> 4) & 31;
        int tile = id >> 9;
        int nt   = tile % (NB_ / 16);
        int kt   = tile / (NB_ / 16);
        int k    = kt * 32 + ((lane & 16) ? 16 : 0) + j;
        int n    = nt * 16 + (lane & 15);
        float v  = 0.0f;
        if (isB) {
            if (k < 300)                   v = W_buf[k * 256 + n];
            else if (k >= 320 && k < 620)  v = W_s1[(k - 320) * 256 + n];
            else if (k >= 640 && k < 940)  v = W_s2[(k - 640) * 256 + n];
            WbSW[id] = (_Float16)v;
        } else {
            if (k < 256) v = W_ih[n * 256 + k];          // W_ih^T
            else         v = W_hh[n * 64 + (k - 256)];   // W_hh^T
            WgSW[id] = (_Float16)v;
        }
    }
}

// Build swizzled f16 weight image for reduce (Wr: 704 x 1536-padded).
__global__ void k_build_wr(const float* W_left, const float* W_right, const float* W_track,
                           _Float16* WrSW) {
    const int total = (KR_ / 32) * (NR_ / 16) * 512;     // 22*96*512 = 1,081,344
    for (int idx = blockIdx.x * 256 + threadIdx.x; idx < total; idx += gridDim.x * 256) {
        int j    = idx & 15;
        int lane = (idx >> 4) & 31;
        int tile = idx >> 9;
        int nt   = tile % (NR_ / 16);
        int kt   = tile / (NR_ / 16);
        int k    = kt * 32 + ((lane & 16) ? 16 : 0) + j;
        int n    = nt * 16 + (lane & 15);
        float v  = 0.0f;
        if (n < NRr) {
            if (k < 300)                   v = W_left[(size_t)k * NRr + n];
            else if (k >= 320 && k < 620)  v = W_right[(size_t)(k - 320) * NRr + n];
            else if (k >= 640 && k < 704)  v = W_track[(k - 640) * NRr + n];
        }
        WrSW[idx] = (_Float16)v;
    }
}

// ------------------------------ step kernels -------------------------------

// KA: lstm_in(f16) = [buf_h | s1_h | s2_h] @ Wb.  grid=(8), block=128 (4 waves).
__global__ void k_bundle(const float* bufs, const float* stack,
                         const int* sptr, const int* bptr,
                         const _Float16* WbSW, _Float16* XgLstm) {
    __shared__ __align__(16) _Float16 Xa[16 * KB_];
    int rt  = blockIdx.x;
    int tid = threadIdx.x;
    for (int i = tid; i < 16 * KB_; i += 128) {
        int r = i / KB_, c = i % KB_;
        int b = rt * 16 + r;
        float v = 0.0f;
        if (c < 320) {
            if (c < Hdim) {
                int bp = bptr[b]; bp = bp < 0 ? 0 : (bp > Sseq - 1 ? Sseq - 1 : bp);
                v = bufs[((size_t)b * Sseq + bp) * D2 + c];
            }
        } else if (c < 640) {
            int cc = c - 320;
            if (cc < Hdim) {
                int sp = sptr[b];
                if (sp >= 1) v = stack[((size_t)b * SLOTS + (sp - 1)) * D2 + cc];
            }
        } else {
            int cc = c - 640;
            if (cc < Hdim) {
                int sp = sptr[b];
                if (sp >= 2) v = stack[((size_t)b * SLOTS + (sp - 2)) * D2 + cc];
            }
        }
        Xa[i] = (_Float16)v;
    }
    __syncthreads();
    int lane = tid & 31, wave = tid >> 5;
    v8f acc0 = {}, acc1 = {}, acc2 = {}, acc3 = {};
#pragma unroll 2
    for (int kt = 0; kt < KB_ / 32; ++kt) {
        v16h a  = load_a_lds(Xa, KB_, kt, lane);
        v16h b0 = load_b_sw(WbSW, NB_ / 16, kt, wave * 4 + 0, lane);
        v16h b1 = load_b_sw(WbSW, NB_ / 16, kt, wave * 4 + 1, lane);
        v16h b2 = load_b_sw(WbSW, NB_ / 16, kt, wave * 4 + 2, lane);
        v16h b3 = load_b_sw(WbSW, NB_ / 16, kt, wave * 4 + 3, lane);
        acc0 = __builtin_amdgcn_wmma_f32_16x16x32_f16(false, a, false, b0, (short)0, acc0, false, false);
        acc1 = __builtin_amdgcn_wmma_f32_16x16x32_f16(false, a, false, b1, (short)0, acc1, false, false);
        acc2 = __builtin_amdgcn_wmma_f32_16x16x32_f16(false, a, false, b2, (short)0, acc2, false, false);
        acc3 = __builtin_amdgcn_wmma_f32_16x16x32_f16(false, a, false, b3, (short)0, acc3, false, false);
    }
    int n0 = (lane & 15);
    int mbase = (lane & 16) ? 8 : 0;
    v8f* accs[4] = { &acc0, &acc1, &acc2, &acc3 };
#pragma unroll
    for (int t = 0; t < 4; ++t) {
        int n = (wave * 4 + t) * 16 + n0;
#pragma unroll
        for (int r = 0; r < 8; ++r)
            XgLstm[(size_t)(rt * 16 + mbase + r) * NB_ + n] = (_Float16)((*accs[t])[r]);
    }
}

// KB: gates GEMM + LSTM cell + transition logits.  grid=(8), block=128.
__global__ void k_tracker(const _Float16* XgLstm, const _Float16* WgSW,
                          float* th, float* tc, _Float16* ThH,
                          const float* W_trans, const float* b_trans,
                          float* out, int step) {
    __shared__ __align__(16) _Float16 Xa[16 * KG_];
    __shared__ float Gs[16 * 256];
    __shared__ float ThS[16 * TRK];
    int rt = blockIdx.x;
    int tid = threadIdx.x;
    for (int i = tid; i < 16 * KG_; i += 128) {
        int r = i / KG_, c = i % KG_;
        int b = rt * 16 + r;
        Xa[i] = (c < 256) ? XgLstm[(size_t)b * 256 + c]
                          : (_Float16)th[b * TRK + (c - 256)];   // th_prev
    }
    __syncthreads();
    int lane = tid & 31, wave = tid >> 5;
    {
        v8f acc0 = {}, acc1 = {}, acc2 = {}, acc3 = {};
#pragma unroll 2
        for (int kt = 0; kt < KG_ / 32; ++kt) {
            v16h a  = load_a_lds(Xa, KG_, kt, lane);
            v16h b0 = load_b_sw(WgSW, NB_ / 16, kt, wave * 4 + 0, lane);
            v16h b1 = load_b_sw(WgSW, NB_ / 16, kt, wave * 4 + 1, lane);
            v16h b2 = load_b_sw(WgSW, NB_ / 16, kt, wave * 4 + 2, lane);
            v16h b3 = load_b_sw(WgSW, NB_ / 16, kt, wave * 4 + 3, lane);
            acc0 = __builtin_amdgcn_wmma_f32_16x16x32_f16(false, a, false, b0, (short)0, acc0, false, false);
            acc1 = __builtin_amdgcn_wmma_f32_16x16x32_f16(false, a, false, b1, (short)0, acc1, false, false);
            acc2 = __builtin_amdgcn_wmma_f32_16x16x32_f16(false, a, false, b2, (short)0, acc2, false, false);
            acc3 = __builtin_amdgcn_wmma_f32_16x16x32_f16(false, a, false, b3, (short)0, acc3, false, false);
        }
        int n0 = (lane & 15);
        int mbase = (lane & 16) ? 8 : 0;
        v8f* accs[4] = { &acc0, &acc1, &acc2, &acc3 };
#pragma unroll
        for (int t = 0; t < 4; ++t) {
            int n = (wave * 4 + t) * 16 + n0;
#pragma unroll
            for (int r = 0; r < 8; ++r) Gs[(mbase + r) * 256 + n] = (*accs[t])[r];
        }
    }
    __syncthreads();
    // LSTM cell: i,f,g,o chunks of 64 (torch order).
    for (int i = tid; i < 16 * TRK; i += 128) {
        int r = i / TRK, t = i % TRK;
        int b = rt * 16 + r;
        float gi = Gs[r * 256 + t],        gf = Gs[r * 256 + 64 + t];
        float gg = Gs[r * 256 + 128 + t],  go = Gs[r * 256 + 192 + t];
        float c_new = sigm(gf) * tc[b * TRK + t] + sigm(gi) * tanhf(gg);
        float h_new = sigm(go) * tanhf(c_new);
        tc[b * TRK + t]  = c_new;
        th[b * TRK + t]  = h_new;
        ThH[b * TRK + t] = (_Float16)h_new;
        ThS[i] = h_new;
    }
    __syncthreads();
    if (tid < 32) {                       // 16 batches x 2 logits
        int r = tid >> 1, j = tid & 1;
        int b = rt * 16 + r;
        float s = b_trans[j];
        for (int t = 0; t < TRK; ++t) s += ThS[r * TRK + t] * W_trans[t * 2 + j];
        out[(size_t)b * OUTCOLS + 300 + 2 * step + j] = s;
    }
}

// KC: a = [s2_h | s1_h | th_new] @ Wr + b_reduce.  grid=(8,6), block=128.
// Each wave owns 4 N-tiles (WG covers 16); Wr is padded to 96 tiles so B
// fragment loads need no guard; stores are guarded at n < 1500.
__global__ void k_reduce_mm(const float* stack, const int* sptr,
                            const _Float16* ThH, const _Float16* WrSW,
                            const float* b_reduce, float* aout) {
    __shared__ __align__(16) _Float16 Xa[16 * KR_];
    int rt = blockIdx.x, cg = blockIdx.y;
    int tid = threadIdx.x;
    for (int i = tid; i < 16 * KR_; i += 128) {
        int r = i / KR_, c = i % KR_;
        int b = rt * 16 + r;
        if (c < 640) {
            float v = 0.0f;
            if (c < 320) {
                if (c < Hdim) {
                    int sp = sptr[b];
                    if (sp >= 2) v = stack[((size_t)b * SLOTS + (sp - 2)) * D2 + c];
                }
            } else {
                int cc = c - 320;
                if (cc < Hdim) {
                    int sp = sptr[b];
                    if (sp >= 1) v = stack[((size_t)b * SLOTS + (sp - 1)) * D2 + cc];
                }
            }
            Xa[i] = (_Float16)v;
        } else {
            Xa[i] = ThH[b * TRK + (c - 640)];
        }
    }
    __syncthreads();
    int lane = tid & 31, wave = tid >> 5;
    int ntb = cg * 16 + wave * 4;        // base N-tile for this wave
    v8f acc0 = {}, acc1 = {}, acc2 = {}, acc3 = {};
#pragma unroll 2
    for (int kt = 0; kt < KR_ / 32; ++kt) {
        v16h a  = load_a_lds(Xa, KR_, kt, lane);
        v16h b0 = load_b_sw(WrSW, NR_ / 16, kt, ntb + 0, lane);
        v16h b1 = load_b_sw(WrSW, NR_ / 16, kt, ntb + 1, lane);
        v16h b2 = load_b_sw(WrSW, NR_ / 16, kt, ntb + 2, lane);
        v16h b3 = load_b_sw(WrSW, NR_ / 16, kt, ntb + 3, lane);
        acc0 = __builtin_amdgcn_wmma_f32_16x16x32_f16(false, a, false, b0, (short)0, acc0, false, false);
        acc1 = __builtin_amdgcn_wmma_f32_16x16x32_f16(false, a, false, b1, (short)0, acc1, false, false);
        acc2 = __builtin_amdgcn_wmma_f32_16x16x32_f16(false, a, false, b2, (short)0, acc2, false, false);
        acc3 = __builtin_amdgcn_wmma_f32_16x16x32_f16(false, a, false, b3, (short)0, acc3, false, false);
    }
    int n0 = (lane & 15);
    int mbase = (lane & 16) ? 8 : 0;
    v8f* accs[4] = { &acc0, &acc1, &acc2, &acc3 };
#pragma unroll
    for (int t = 0; t < 4; ++t) {
        int n = (ntb + t) * 16 + n0;
        if (n < NRr) {
            float bias = b_reduce[n];
#pragma unroll
            for (int r = 0; r < 8; ++r)
                aout[(size_t)(rt * 16 + mbase + r) * NRr + n] = (*accs[t])[r] + bias;
        }
    }
}

// KD: TreeLSTM reduce elementwise + shift/reduce stack update. grid=16, block=256.
__global__ void k_step_update(const float* bufs, const int* transitions,
                              const float* aout, float* stack,
                              int* sptr, int* bptr, int step) {
    int wave = threadIdx.x >> 5;
    int lane = threadIdx.x & 31;
    int b = blockIdx.x * 8 + wave;
    int sp = sptr[b];
    int bp = bptr[b];
    int bpc = bp < 0 ? 0 : (bp > Sseq - 1 ? Sseq - 1 : bp);
    int trans = transitions[b * NSTEPS + step];
    bool shift = (trans == 0);
    int wpos = shift ? sp : (sp - 2 < 0 ? 0 : sp - 2);
    int p1 = sp - 1 < 0 ? 0 : sp - 1;
    int p2 = sp - 2 < 0 ? 0 : sp - 2;
    const float* arow = aout + (size_t)b * NRr;
    for (int h = lane; h < Hdim; h += 32) {
        float ri = arow[h],        fl = arow[300 + h], fr = arow[600 + h];
        float ro = arow[900 + h],  rg = arow[1200 + h];
        float s1c = (sp >= 1) ? stack[((size_t)b * SLOTS + p1) * D2 + Hdim + h] : 0.0f;
        float s2c = (sp >= 2) ? stack[((size_t)b * SLOTS + p2) * D2 + Hdim + h] : 0.0f;
        float rc = sigm(fl) * s2c + sigm(fr) * s1c + sigm(ri) * tanhf(rg);
        float rh = sigm(ro) * tanhf(rc);
        float bh = bufs[((size_t)b * Sseq + bpc) * D2 + h];
        float bc = bufs[((size_t)b * Sseq + bpc) * D2 + Hdim + h];
        stack[((size_t)b * SLOTS + wpos) * D2 + h]        = shift ? bh : rh;
        stack[((size_t)b * SLOTS + wpos) * D2 + Hdim + h] = shift ? bc : rc;
    }
    if (lane == 0) {
        sptr[b] = sp + (shift ? 1 : -1);
        bptr[b] = bp + (shift ? 1 : 0);
    }
}

__global__ void k_final(const float* stack, const int* sptr, float* out) {
    int idx = blockIdx.x * 256 + threadIdx.x;
    if (idx >= Bsz * Hdim) return;
    int b = idx / Hdim, h = idx % Hdim;
    int sp = sptr[b];
    int p = sp - 1 < 0 ? 0 : sp - 1;
    out[(size_t)b * OUTCOLS + h] = stack[((size_t)b * SLOTS + p) * D2 + h];
}

// ------------------------------- launcher ----------------------------------

extern "C" void kernel_launch(void* const* d_in, const int* in_sizes, int n_in,
                              void* d_out, int out_size, void* d_ws, size_t ws_size,
                              hipStream_t stream) {
    const float* bufs       = (const float*)d_in[0];
    const int*   transitions= (const int*)d_in[1];
    const float* W_buf      = (const float*)d_in[2];
    const float* W_s1       = (const float*)d_in[3];
    const float* W_s2       = (const float*)d_in[4];
    const float* W_ih       = (const float*)d_in[5];
    const float* W_hh       = (const float*)d_in[6];
    const float* W_trans    = (const float*)d_in[7];
    const float* b_trans    = (const float*)d_in[8];
    const float* W_left     = (const float*)d_in[9];
    const float* W_right    = (const float*)d_in[10];
    const float* W_track    = (const float*)d_in[11];
    const float* b_reduce   = (const float*)d_in[12];
    float* out = (float*)d_out;

    char* ws = (char*)d_ws;
    size_t off = 0;
    auto alloc = [&](size_t bytes) -> void* {
        void* p = ws + off;
        off = (off + bytes + 255) & ~(size_t)255;
        return p;
    };
    float*     stack  = (float*)alloc((size_t)Bsz * SLOTS * D2 * sizeof(float));   // ~79 MB
    int*       sptr   = (int*)alloc(Bsz * sizeof(int));
    int*       bptr   = (int*)alloc(Bsz * sizeof(int));
    float*     th     = (float*)alloc(Bsz * TRK * sizeof(float));
    float*     tc     = (float*)alloc(Bsz * TRK * sizeof(float));
    _Float16*  ThH    = (_Float16*)alloc(Bsz * TRK * sizeof(_Float16));
    _Float16*  XgLstm = (_Float16*)alloc((size_t)Bsz * NB_ * sizeof(_Float16));
    _Float16*  WbSW   = (_Float16*)alloc((size_t)KB_ * NB_ * sizeof(_Float16));
    _Float16*  WgSW   = (_Float16*)alloc((size_t)KG_ * NB_ * sizeof(_Float16));
    _Float16*  WrSW   = (_Float16*)alloc((size_t)KR_ * NR_ * sizeof(_Float16));
    float*     aout   = (float*)alloc((size_t)Bsz * NRr * sizeof(float));
    (void)in_sizes; (void)n_in; (void)out_size; (void)ws_size;

    k_init_state<<<32, 256, 0, stream>>>(sptr, bptr, th, tc);
    k_build_wbg<<<512, 256, 0, stream>>>(W_buf, W_s1, W_s2, W_ih, W_hh, WbSW, WgSW);
    k_build_wr<<<1024, 256, 0, stream>>>(W_left, W_right, W_track, WrSW);

    for (int step = 0; step < NSTEPS; ++step) {
        k_bundle<<<8, 128, 0, stream>>>(bufs, stack, sptr, bptr, WbSW, XgLstm);
        k_tracker<<<8, 128, 0, stream>>>(XgLstm, WgSW, th, tc, ThH,
                                         W_trans, b_trans, out, step);
        k_reduce_mm<<<dim3(8, 6), 128, 0, stream>>>(stack, sptr, ThH, WrSW,
                                                    b_reduce, aout);
        k_step_update<<<16, 256, 0, stream>>>(bufs, transitions, aout, stack,
                                              sptr, bptr, step);
    }
    k_final<<<(Bsz * Hdim + 255) / 256, 256, 0, stream>>>(stack, sptr, out);
}
